// LocalMessagePassing_86053964743173
// MI455X (gfx1250) — compile-verified, matrix operand order checked by reading, not run on hardware
//
#include <hip/hip_runtime.h>
#include <hip/hip_bf16.h>
#include <math.h>

// ---------------- problem constants (fixed shapes from setup_inputs) --------
#define S_SPEC 4
#define B_MOL  256
#define A_ATOM 128
#define N_ATOM (B_MOL * A_ATOM)        // 32768
#define AEV_D  1008
#define MO_D   256
#define NO_D   128
#define FIN_D  (MO_D + NO_D)           // 384
#define P_PAIR 262144
#define CUTOFF_F 5.2f

typedef __attribute__((ext_vector_type(16))) __bf16 v16bf;
typedef __attribute__((ext_vector_type(4)))  __bf16 v4bf;
typedef __attribute__((ext_vector_type(8)))  float  v8f;

static __device__ __forceinline__ __bf16 tobf(float x) { return (__bf16)x; }

// K index inside a 32-wide chunk for fragment element e (0..15), lane group g (0/1)
// per CDNA5 16-bit A/B fragment layout (ISA 7.12.2).
static __device__ __forceinline__ int kOf(int e, int g) {
  return e + 8 * g + (e >= 8 ? 8 : 0);
}

static __device__ __forceinline__ float gelu_f(float x) {
  float x3 = x * x * x;
  return 0.5f * x * (1.0f + tanhf(0.7978845608028654f * (x + 0.044715f * x3)));
}

// ---------------- 1) species compaction (deterministic, 1 block) ------------
__global__ void compact_kernel(const int* __restrict__ species,
                               int* __restrict__ perm, int* __restrict__ offs) {
  int lane = threadIdx.x & 31;
  int w = threadIdx.x >> 5;  // wave w handles species w (4 waves)
  __shared__ int counts[S_SPEC];
  int cnt = 0;
  for (int i = lane; i < N_ATOM; i += 32)
    if (species[i] == w) cnt++;
  for (int off = 16; off; off >>= 1) cnt += __shfl_down(cnt, off, 32);
  if (lane == 0) counts[w] = cnt;
  __syncthreads();
  if (threadIdx.x == 0) {
    int acc = 0;
    offs[0] = 0;
    for (int sp = 0; sp < S_SPEC; sp++) { acc += counts[sp]; offs[sp + 1] = acc; }
  }
  int base = 0;
  for (int sp = 0; sp < w; sp++) base += counts[sp];
  // stable append (order = atom index order within species)
  int pos = base;
  for (int i0 = 0; i0 < N_ATOM; i0 += 32) {
    int i = i0 + lane;
    bool m = species[i] == w;
    unsigned long long mk = __ballot(m);
    int rank = __popcll(mk & ((1ull << lane) - 1ull));
    if (m) perm[pos + rank] = i;
    pos += __popcll(mk);
  }
}

// ---------------- 2) decay = pf^2 * exp(-fa^2 d) * smooth_cutoff(d) ---------
__global__ void decay_kernel(const float* __restrict__ dist,
                             const float* __restrict__ factor,
                             const float* __restrict__ prefactor,
                             float* __restrict__ decay) {
  int p = blockIdx.x * blockDim.x + threadIdx.x;
  if (p >= P_PAIR) return;
  float fa = factor[0], pf = prefactor[0];
  float d = dist[p];
  float x = fminf(fmaxf(d / CUTOFF_F, 0.0f), 1.0f - 1e-6f);
  float f = __expf(1.0f - 1.0f / (1.0f - x * x));
  float cut = (d < CUTOFF_F) ? f : 0.0f;
  decay[p] = pf * pf * __expf(-fa * fa * d) * cut;
}

// ---------------- 3) weight pre-pack into WMMA B-fragment order (bf16) ------
// packed[((s*nct + ct)*nch + c)*32 + lane][e] = W[s][c*32 + kOf(e,g)][ct*16 + col]
__global__ void pack_w_kernel(const float* __restrict__ W, __bf16* __restrict__ out,
                              int K, int Nout, int nct, int nch) {
  int tid = blockIdx.x * blockDim.x + threadIdx.x;
  int total = S_SPEC * nct * nch * 32;
  if (tid >= total) return;
  int lane = tid & 31;
  int r = tid >> 5;
  int c = r % nch; r /= nch;
  int ct = r % nct;
  int s = r / nct;
  int g = lane >> 4;
  int col = (lane & 15) + ct * 16;
  __bf16* dst = out + ((((size_t)s * nct + ct) * nch + c) * 32 + lane) * 16;
#pragma unroll
  for (int e = 0; e < 16; e++) {
    int k = c * 32 + kOf(e, g);
    float v = (k < K) ? W[((size_t)s * K + k) * Nout + col] : 0.0f;
    dst[e] = tobf(v);
  }
}

// ---------------- 4) species-routed WMMA GEMM -------------------------------
// block = 128 threads (4 waves). Each wave computes RT row-tiles x CT col-tiles
// of 16x16; block covers RT*16 rows x 4*CT*16 cols (single species per block).
// B fragments are loaded once per col-tile per chunk and reused across the RT
// row fragments. A tile is double-buffered in LDS; staging of chunk c+1
// overlaps the WMMAs of chunk c. B addressing is kept as (global base +
// loop-carried offset) so the compiler emits global_load_b128, not flat.
template <int RT, int CT, bool GELU>
__global__ void gemm_routed(const float* __restrict__ in, int inStride,
                            float* __restrict__ out, int outStride, int outOffset,
                            const __bf16* __restrict__ packed,
                            const float* __restrict__ bias,
                            const int* __restrict__ perm,
                            const int* __restrict__ offs,
                            int Kin, int Nout) {
  const int nch = (Kin + 31) >> 5;
  const int nct = Nout >> 4;
  const int lane = threadIdx.x & 31;
  const int wave = threadIdx.x >> 5;
  const int ROWS = RT * 16;

  __shared__ int soff[S_SPEC + 1];
  if (threadIdx.x <= S_SPEC) soff[threadIdx.x] = offs[threadIdx.x];
  __syncthreads();

  // map linear block id -> (species, local row super-tile)
  int t = blockIdx.x;
  int s = -1, rowbase = 0;
  for (int sp = 0; sp < S_SPEC; sp++) {
    int tiles = (soff[sp + 1] - soff[sp] + ROWS - 1) / ROWS;
    if (t < tiles) { s = sp; rowbase = soff[sp] + t * ROWS; break; }
    t -= tiles;
  }
  if (s < 0) return;
  const int send = soff[s + 1];

  __shared__ __bf16 lA[2][RT * 16 * 32];

  // staging: thread stages RT float4s (row m of each row-tile, k-sub kk..kk+3)
  const int m_st  = threadIdx.x >> 3;        // 0..15
  const int kk_st = (threadIdx.x & 7) << 2;  // 0,4,..,28
  const float* rptr[RT];
#pragma unroll
  for (int rr = 0; rr < RT; rr++) {
    int slot = rowbase + rr * 16 + m_st;
    rptr[rr] = (slot < send) ? in + (size_t)perm[slot] * inStride + kk_st : nullptr;
  }

  auto stage = [&](int c, int buf) {
    int k = c * 32 + kk_st;
    bool kok = (k + 4) <= Kin;
#pragma unroll
    for (int rr = 0; rr < RT; rr++) {
      v4bf bv = {};
      if (rptr[rr] != nullptr && kok) {
        float4 v = *(const float4*)(rptr[rr] + c * 32);
        bv[0] = tobf(v.x); bv[1] = tobf(v.y); bv[2] = tobf(v.z); bv[3] = tobf(v.w);
      }
      *(v4bf*)(&lA[buf][rr * 512 + (m_st << 5) + kk_st]) = bv;
    }
  };

  v8f acc[RT][CT];
#pragma unroll
  for (int rr = 0; rr < RT; rr++)
#pragma unroll
    for (int tt = 0; tt < CT; tt++) acc[rr][tt] = (v8f){};

  // B fragment addressing: global base + loop-carried element offset
  const size_t tileStride = (size_t)nch * 512;   // bf16 elems per col-tile
  size_t boff = ((size_t)s * nct + wave * CT) * tileStride + lane * 16;

  const int g = lane >> 4;
  const int mrow = lane & 15;

  stage(0, 0);
  __syncthreads();
  for (int c = 0; c < nch; c++) {
    const int buf = c & 1;
    if (c + 1 < nch) stage(c + 1, buf ^ 1);  // overlaps WMMAs below
    v16bf a[RT];
#pragma unroll
    for (int rr = 0; rr < RT; rr++)
#pragma unroll
      for (int e = 0; e < 16; e++)
        a[rr][e] = lA[buf][rr * 512 + (mrow << 5) + kOf(e, g)];
#pragma unroll
    for (int tt = 0; tt < CT; tt++) {
      v16bf b = *(const v16bf*)(packed + boff + tt * tileStride);
#pragma unroll
      for (int rr = 0; rr < RT; rr++)
        acc[rr][tt] = __builtin_amdgcn_wmma_f32_16x16x32_bf16(
            false, a[rr], false, b, (short)0, acc[rr][tt], false, false);
    }
    boff += 512;
    __syncthreads();
  }

  // epilogue: bias + (gelu) + scatter back to original atom rows
#pragma unroll
  for (int tt = 0; tt < CT; tt++) {
    int col = (wave * CT + tt) * 16 + (lane & 15);
    float bv = bias[s * Nout + col];
#pragma unroll
    for (int rr = 0; rr < RT; rr++) {
#pragma unroll
      for (int r = 0; r < 8; r++) {
        int slot = rowbase + rr * 16 + r + 8 * g;
        if (slot < send) {
          int row = perm[slot];
          float v = acc[rr][tt][r] + bv;
          if (GELU) v = gelu_f(v);
          out[(size_t)row * outStride + outOffset + col] = v;
        }
      }
    }
  }
}

// ---------------- 5) zero strided column block ------------------------------
__global__ void zero_cols_kernel(float* __restrict__ out, int stride, int offset,
                                 int ncols) {
  int tid = blockIdx.x * blockDim.x + threadIdx.x;
  int total = N_ATOM * ncols;
  if (tid >= total) return;
  int row = tid / ncols, c = tid - row * ncols;
  out[(size_t)row * stride + offset + c] = 0.0f;
}

// ---------------- 6) pair message scatter (both directions) -----------------
__global__ void message_kernel(const float* __restrict__ neigh,
                               const int* __restrict__ idx,   // [2, P]
                               const float* __restrict__ decay,
                               float* __restrict__ out, int outStride, int outOffset) {
  int p = blockIdx.x * (blockDim.x >> 5) + (threadIdx.x >> 5);
  int lane = threadIdx.x & 31;
  if (p >= P_PAIR) return;
  int i0 = idx[p];
  int i1 = idx[P_PAIR + p];
  float d = decay[p];
  float4 n0 = ((const float4*)(neigh + (size_t)i0 * NO_D))[lane];
  float4 n1 = ((const float4*)(neigh + (size_t)i1 * NO_D))[lane];
  float* o0 = out + (size_t)i0 * outStride + outOffset + lane * 4;
  float* o1 = out + (size_t)i1 * outStride + outOffset + lane * 4;
  // target i0 <- neigh[i1]*d ; target i1 <- neigh[i0]*d
  atomicAdd(o0 + 0, n1.x * d); atomicAdd(o0 + 1, n1.y * d);
  atomicAdd(o0 + 2, n1.z * d); atomicAdd(o0 + 3, n1.w * d);
  atomicAdd(o1 + 0, n0.x * d); atomicAdd(o1 + 1, n0.y * d);
  atomicAdd(o1 + 2, n0.z * d); atomicAdd(o1 + 3, n0.w * d);
}

// ---------------- 7) routed linear head: pre = feat . Wf[s] + bf[s] ---------
__global__ void final_kernel(const float* __restrict__ feat,
                             const float* __restrict__ Wf,
                             const float* __restrict__ bf,
                             const int* __restrict__ species,
                             float* __restrict__ pre_out) {
  int n = blockIdx.x * (blockDim.x >> 5) + (threadIdx.x >> 5);
  int lane = threadIdx.x & 31;
  if (n >= N_ATOM) return;
  int s = species[n];
  const float4* f4 = (const float4*)(feat + (size_t)n * FIN_D);
  const float4* w4 = (const float4*)(Wf + (size_t)s * FIN_D);
  float acc = 0.0f;
#pragma unroll
  for (int j = 0; j < FIN_D / 128; j++) {  // 3 coalesced float4 per lane
    float4 f = f4[lane + 32 * j];
    float4 w = w4[lane + 32 * j];
    acc += f.x * w.x + f.y * w.y + f.z * w.z + f.w * w.w;
  }
  for (int off = 16; off; off >>= 1) acc += __shfl_down(acc, off, 32);
  if (lane == 0) pre_out[n] = acc + bf[s];
}

// ---------------- 8) per-molecule charge redistribution ---------------------
__global__ void charge_kernel(const int* __restrict__ species,
                              const float* __restrict__ total_charges,
                              const float* __restrict__ pre,
                              float* __restrict__ out_charges,
                              int* __restrict__ out_species) {
  int b = blockIdx.x;        // molecule
  int a = threadIdx.x;       // atom (A_ATOM = 128 threads = 4 waves)
  int i = b * A_ATOM + a;
  float p = pre[i];
  int sp = species[i];
  bool real = (sp != -1);
  float v = real ? p : 0.0f;
  float c = real ? 1.0f : 0.0f;
  for (int off = 16; off; off >>= 1) {
    v += __shfl_down(v, off, 32);
    c += __shfl_down(c, off, 32);
  }
  __shared__ float ssum[4], scnt[4];
  int lane = a & 31, w = a >> 5;
  if (lane == 0) { ssum[w] = v; scnt[w] = c; }
  __syncthreads();
  float sum = ssum[0] + ssum[1] + ssum[2] + ssum[3];
  float nreal = scnt[0] + scnt[1] + scnt[2] + scnt[3];
  float q = total_charges[b];
  float ch = p + (q - sum) / nreal;
  out_charges[i] = real ? ch : 0.0f;
  out_species[i] = sp;
}

// ---------------- launch ----------------------------------------------------
extern "C" void kernel_launch(void* const* d_in, const int* in_sizes, int n_in,
                              void* d_out, int out_size, void* d_ws, size_t ws_size,
                              hipStream_t stream) {
  const int*   species   = (const int*)d_in[0];
  const float* aev       = (const float*)d_in[1];
  const int*   idx12     = (const int*)d_in[2];
  const float* dist      = (const float*)d_in[3];
  const float* qtot      = (const float*)d_in[4];
  const float* Wm0 = (const float*)d_in[5];  const float* bm0 = (const float*)d_in[6];
  const float* Wn0 = (const float*)d_in[7];  const float* bn0 = (const float*)d_in[8];
  const float* Wm1 = (const float*)d_in[9];  const float* bm1 = (const float*)d_in[10];
  const float* Wn1 = (const float*)d_in[11]; const float* bn1 = (const float*)d_in[12];
  const float* Wf  = (const float*)d_in[13]; const float* bf  = (const float*)d_in[14];
  const float* factor    = (const float*)d_in[15];
  const float* prefactor = (const float*)d_in[16];

  // ---- workspace carve-up (256B aligned) ----
  char* ws = (char*)d_ws;
  size_t o = 0;
  auto carve = [&](size_t bytes) { char* p = ws + o; o += (bytes + 255) & ~(size_t)255; return p; };
  int*    perm  = (int*)carve(N_ATOM * sizeof(int));
  int*    offs  = (int*)carve(8 * sizeof(int));
  float*  decay = (float*)carve((size_t)P_PAIR * sizeof(float));
  float*  feat1 = (float*)carve((size_t)N_ATOM * FIN_D * sizeof(float));
  float*  feat2 = (float*)carve((size_t)N_ATOM * FIN_D * sizeof(float));
  float*  neigh = (float*)carve((size_t)N_ATOM * NO_D * sizeof(float));
  // packed bf16 weights (K padded up to multiple of 32)
  const int nct_m0 = MO_D / 16, nch_m0 = (AEV_D + 31) / 32;   // 16, 32
  const int nct_n  = NO_D / 16, nch_n  = MO_D / 32;           // 8, 8
  const int nct_m1 = MO_D / 16, nch_m1 = FIN_D / 32;          // 16, 12
  __bf16* pWm0 = (__bf16*)carve((size_t)S_SPEC * nct_m0 * nch_m0 * 512 * sizeof(__bf16));
  __bf16* pWn0 = (__bf16*)carve((size_t)S_SPEC * nct_n  * nch_n  * 512 * sizeof(__bf16));
  __bf16* pWm1 = (__bf16*)carve((size_t)S_SPEC * nct_m1 * nch_m1 * 512 * sizeof(__bf16));
  __bf16* pWn1 = (__bf16*)carve((size_t)S_SPEC * nct_n  * nch_n  * 512 * sizeof(__bf16));

  float* out_f    = (float*)d_out;
  int*   out_sp   = (int*)d_out;             // segment 0: species (int32 bits)
  float* out_chg  = out_f + N_ATOM;          // segment 1: charges
  float* out_pre  = out_f + 2 * N_ATOM;      // segment 2: pre

  const int GEMM_GRID = N_ATOM / 32 + S_SPEC;  // RT=2 super-tiles, worst case

  // 1) compaction + decay + weight packing
  compact_kernel<<<1, 128, 0, stream>>>(species, perm, offs);
  decay_kernel<<<P_PAIR / 256, 256, 0, stream>>>(dist, factor, prefactor, decay);
  {
    int t0 = S_SPEC * nct_m0 * nch_m0 * 32;
    pack_w_kernel<<<(t0 + 255) / 256, 256, 0, stream>>>(Wm0, pWm0, AEV_D, MO_D, nct_m0, nch_m0);
    int t1 = S_SPEC * nct_n * nch_n * 32;
    pack_w_kernel<<<(t1 + 255) / 256, 256, 0, stream>>>(Wn0, pWn0, MO_D, NO_D, nct_n, nch_n);
    int t2 = S_SPEC * nct_m1 * nch_m1 * 32;
    pack_w_kernel<<<(t2 + 255) / 256, 256, 0, stream>>>(Wm1, pWm1, FIN_D, MO_D, nct_m1, nch_m1);
    pack_w_kernel<<<(t1 + 255) / 256, 256, 0, stream>>>(Wn1, pWn1, MO_D, NO_D, nct_n, nch_n);
  }

  // 2) round 0: internal -> feat1[:, :256], neigh, message -> feat1[:, 256:]
  gemm_routed<2, 4, true><<<GEMM_GRID, 128, 0, stream>>>(
      aev, AEV_D, feat1, FIN_D, 0, pWm0, bm0, perm, offs, AEV_D, MO_D);
  gemm_routed<2, 2, true><<<GEMM_GRID, 128, 0, stream>>>(
      feat1, FIN_D, neigh, NO_D, 0, pWn0, bn0, perm, offs, MO_D, NO_D);
  zero_cols_kernel<<<(N_ATOM * NO_D + 255) / 256, 256, 0, stream>>>(feat1, FIN_D, MO_D, NO_D);
  message_kernel<<<P_PAIR / 8, 256, 0, stream>>>(neigh, idx12, decay, feat1, FIN_D, MO_D);

  // 3) round 1: feat1 -> feat2[:, :256], neigh, message -> feat2[:, 256:]
  gemm_routed<2, 4, true><<<GEMM_GRID, 128, 0, stream>>>(
      feat1, FIN_D, feat2, FIN_D, 0, pWm1, bm1, perm, offs, FIN_D, MO_D);
  gemm_routed<2, 2, true><<<GEMM_GRID, 128, 0, stream>>>(
      feat2, FIN_D, neigh, NO_D, 0, pWn1, bn1, perm, offs, MO_D, NO_D);
  zero_cols_kernel<<<(N_ATOM * NO_D + 255) / 256, 256, 0, stream>>>(feat2, FIN_D, MO_D, NO_D);
  message_kernel<<<P_PAIR / 8, 256, 0, stream>>>(neigh, idx12, decay, feat2, FIN_D, MO_D);

  // 4) head + charge redistribution
  final_kernel<<<N_ATOM / 8, 256, 0, stream>>>(feat2, Wf, bf, species, out_pre);
  charge_kernel<<<B_MOL, A_ATOM, 0, stream>>>(species, qtot, out_pre, out_chg, out_sp);

  (void)in_sizes; (void)n_in; (void)out_size; (void)ws_size;
}
